// Transducer_33595234189449
// MI455X (gfx1250) — compile-verified, matrix operand order checked by reading, not run on hardware
//
#include <hip/hip_runtime.h>
#include <hip/hip_bf16.h>
#include <stdint.h>

// ---------------- constants (match setup_inputs) ----------------
#define BB   16            // batch
#define TT   1024          // time frames
#define EE   512           // encoder dim
#define DD   512           // decoder dim
#define JJ   512           // joiner dim
#define VV   500           // vocab
#define NP   512           // padded vocab / universal N
#define PP   256           // num_paths
#define LL   32            // path_length
#define CTX  2
#define NROWS (BB*PP)      // 4096

typedef __attribute__((ext_vector_type(16))) __bf16 v16bf;
typedef __attribute__((ext_vector_type(8)))  float  v8f;
typedef __attribute__((ext_vector_type(4)))  unsigned int v4u;
typedef __attribute__((ext_vector_type(8)))  int v8i;
typedef __attribute__((ext_vector_type(4)))  int v4i;

#if defined(__gfx1250__) && __has_builtin(__builtin_amdgcn_tensor_load_to_lds) && __has_builtin(__builtin_amdgcn_s_wait_tensorcnt)
#define HAVE_TDM 1
#else
#define HAVE_TDM 0
#endif

union Frag  { v16bf v; unsigned int u[8]; };
union Q16   { uint4 q; unsigned short s[8]; };

__device__ __forceinline__ unsigned short f32_to_bf16(float f) {
    unsigned int u = __float_as_uint(f);
    unsigned int r = u + 0x7fffu + ((u >> 16) & 1u);   // round-to-nearest-even
    return (unsigned short)(r >> 16);
}

__device__ __forceinline__ unsigned long long splitmix64(unsigned long long x) {
    x += 0x9E3779B97F4A7C15ull;
    x = (x ^ (x >> 30)) * 0xBF58476D1CE4E5B9ull;
    x = (x ^ (x >> 27)) * 0x94D049BB133111EBull;
    return x ^ (x >> 31);
}

// ---------------- init: path state + min_tmax ----------------
__global__ void init_kernel(const int* lens, int* t_index, int* left, int* minTmax) {
    int tid = blockIdx.x * blockDim.x + threadIdx.x;
    if (tid < NROWS) {
        t_index[tid] = 0;
        left[2 * tid + 0] = 0;
        left[2 * tid + 1] = 0;
    }
    if (tid == 0) {
        int mn = lens[0];
        for (int i = 1; i < BB; ++i) mn = min(mn, lens[i]);
        minTmax[0] = max(mn - 1, 1);
    }
}

// ---------------- weight fp32 -> bf16 (with column padding) ----------------
__global__ void convert_kernel(const float* __restrict__ src, unsigned short* __restrict__ dst,
                               int rows, int cols, int dstCols) {
    int i = blockIdx.x * blockDim.x + threadIdx.x;
    int total = rows * dstCols;
    if (i >= total) return;
    int r = i / dstCols, c = i % dstCols;
    dst[i] = (c < cols) ? f32_to_bf16(src[(size_t)r * cols + c]) : (unsigned short)0;
}

// ---------------- per-step gathers: enc frame + decoder embeddings -> bf16 ----------------
__global__ void gather_kernel(const float* __restrict__ enc, const int* __restrict__ t_index,
                              const int* __restrict__ left,
                              const float* __restrict__ emb_p, const float* __restrict__ emb_h,
                              unsigned short* __restrict__ encX,
                              unsigned short* __restrict__ Xp, unsigned short* __restrict__ Xh) {
    int row = blockIdx.x;                 // 0..NROWS-1
    int b = row >> 8;                     // PP = 256
    int t = t_index[row];
    const float* esrc = enc + ((size_t)b * TT + t) * EE;
    for (int c = threadIdx.x; c < EE; c += blockDim.x)
        encX[(size_t)row * EE + c] = f32_to_bf16(esrc[c]);
    int l0 = left[2 * row + 0], l1 = left[2 * row + 1];
    const float* p0 = emb_p + (size_t)l0 * DD;
    const float* p1 = emb_p + (size_t)l1 * DD;
    const float* h0 = emb_h + (size_t)l0 * DD;
    const float* h1 = emb_h + (size_t)l1 * DD;
    unsigned short* xp = Xp + (size_t)row * (CTX * DD);
    unsigned short* xh = Xh + (size_t)row * (CTX * DD);
    for (int c = threadIdx.x; c < DD; c += blockDim.x) {
        xp[c]      = f32_to_bf16(p0[c]);
        xp[DD + c] = f32_to_bf16(p1[c]);
        xh[c]      = f32_to_bf16(h0[c]);
        xh[DD + c] = f32_to_bf16(h1[c]);
    }
}

// ---------------- generic WMMA bf16 GEMM ----------------
// C(MxN) = epi( A0(MxK0)*B0(K0xN) [+ A1(MxK1)*B1(K1xN)] + bias )
// Tile 128x64, 128 threads (4 waves); each wave -> 32 rows x 64 cols via
// 8x v_wmma_f32_16x16x32_bf16 per K-chunk. A tiles staged by TDM
// (tensor_load_to_lds, double buffered, LDS pad->stride 34); B tiles staged
// manually with reg-batched B128 loads + transpose scatter.
#define BM 128
#define BN 64
#define BK 32
#define LSTR 34

struct Chunk { const unsigned short* A; const unsigned short* B; int K; int kb; };

__device__ __forceinline__ Chunk get_chunk(int c, int n0,
                                           const unsigned short* A0, const unsigned short* B0, int K0,
                                           const unsigned short* A1, const unsigned short* B1, int K1) {
    Chunk s;
    if (c < n0) { s.A = A0; s.B = B0; s.K = K0; s.kb = c * BK; }
    else        { s.A = A1; s.B = B1; s.K = K1; s.kb = (c - n0) * BK; }
    return s;
}

// Issue TDM: load BM x BK bf16 tile (row-major, row stride K elements) into LDS
// at byte offset ldsOff, padding +1 dword per 16 dwords -> LDS row stride 34 ushorts.
__device__ __forceinline__ void tdm_load_A(const unsigned short* A, int K, int mbase, int kb,
                                           unsigned ldsOff) {
#if HAVE_TDM
    unsigned long long ga = (unsigned long long)(const void*)(A + (size_t)mbase * K + kb);
    v4u g0;
    g0[0] = 1u;                                       // count=1 (valid), user mode
    g0[1] = ldsOff;                                   // lds_addr (bytes)
    g0[2] = (unsigned)(ga & 0xffffffffu);             // global_addr[31:0]
    g0[3] = (unsigned)((ga >> 32) & 0x1ffffffu) | (2u << 30); // addr[56:32] | type=2
    v8i g1;
    // data_size=1(2B) | pad_enable | pad_interval=3 (16 dwords) | pad_amount=0 (1 dword)
    g1[0] = (int)(0x00010000u | (1u << 20) | (3u << 22));
    g1[1] = (int)(((unsigned)K & 0xffffu) << 16);                      // tensor_dim0 lo16
    g1[2] = (int)(((unsigned)K >> 16) | (((unsigned)NROWS & 0xffffu) << 16)); // dim0 hi | dim1 lo
    g1[3] = (int)((((unsigned)NROWS >> 16) & 0xffffu) | ((unsigned)BK << 16)); // dim1 hi | tile_dim0=32
    g1[4] = (int)((unsigned)BM);                                       // tile_dim1=128, tile_dim2=0
    g1[5] = (int)(unsigned)K;                                          // tensor_dim0_stride lo32
    g1[6] = 0;                                                         // stride0 hi | stride1 lo
    g1[7] = 0;
    v4i gz4 = {0, 0, 0, 0};
    v8i gz8 = {0, 0, 0, 0, 0, 0, 0, 0};
    __builtin_amdgcn_tensor_load_to_lds(g0, g1, gz4, gz4, gz8, 0);
#else
    (void)A; (void)K; (void)mbase; (void)kb; (void)ldsOff;
#endif
}

__device__ __forceinline__ void stage_A_manual(unsigned short* dst, const unsigned short* A,
                                               int K, int mbase, int kb, int tid) {
    #pragma unroll
    for (int i = 0; i < 16; ++i) {          // 128x32 ushorts = 2048 dwords / 128 thr
        int d = tid + i * 128;
        int row = d >> 4;
        int c2 = (d & 15) << 1;
        unsigned int val = *(const unsigned int*)(A + (size_t)(mbase + row) * K + kb + c2);
        *(unsigned int*)(&dst[row * LSTR + c2]) = val;
    }
}

__global__ void wmma_gemm_kernel(const unsigned short* __restrict__ A0,
                                 const unsigned short* __restrict__ B0, int K0,
                                 const unsigned short* __restrict__ A1,
                                 const unsigned short* __restrict__ B1, int K1,
                                 const float* __restrict__ bias, int biasN,
                                 int M, int N,
                                 unsigned short* __restrict__ outBf,
                                 float* __restrict__ outF, int epilogue) {
    __shared__ unsigned short lA[2][BM * LSTR];   // 2 x 8704 B (TDM double buffer)
    __shared__ unsigned short lB[2][BN * LSTR];   // 2 x 4352 B
    const int mbase = blockIdx.x * BM;
    const int nbase = blockIdx.y * BN;
    const int tid  = threadIdx.x;       // 0..127
    const int wave = tid >> 5;          // 0..3
    const int lane = tid & 31;

    const int n0 = K0 / BK;
    const int n1 = K1 / BK;
    const int nch = n0 + n1;

    unsigned ldsA[2];
    ldsA[0] = (unsigned)(size_t)(void*)&lA[0][0];
    ldsA[1] = (unsigned)(size_t)(void*)&lA[1][0];

    v8f acc[2][4];
    for (int a = 0; a < 2; ++a)
        for (int i = 0; i < 4; ++i)
            for (int j = 0; j < 8; ++j) acc[a][i][j] = 0.f;

#if HAVE_TDM
    if (wave == 0) {               // preload chunk 0's A tile via TDM
        Chunk s = get_chunk(0, n0, A0, B0, K0, A1, B1, K1);
        tdm_load_A(s.A, s.K, mbase, s.kb, ldsA[0]);
    }
#endif

    for (int c = 0; c < nch; ++c) {
        const int cur = c & 1;
        const int nxt = cur ^ 1;
        Chunk s = get_chunk(c, n0, A0, B0, K0, A1, B1, K1);

        // ---- stage B tile: regs first (batched B128 loads), then transpose scatter ----
        Q16 bq[2];
        int kIdx[2], n8Idx[2];
        #pragma unroll
        for (int i = 0; i < 2; ++i) {
            int ci = tid + i * 128;            // 0..255 uint4 chunks
            kIdx[i]  = ci >> 3;                // 0..31
            n8Idx[i] = (ci & 7) << 3;          // 0,8,..,56
            bq[i].q = *(const uint4*)(s.B + (size_t)(s.kb + kIdx[i]) * N + nbase + n8Idx[i]);
        }
        // prefetch next chunk's B tile (global_prefetch_b8)
        if (c + 1 < nch) {
            Chunk t = get_chunk(c + 1, n0, A0, B0, K0, A1, B1, K1);
            const char* pf = (const char*)(t.B + (size_t)(t.kb + (tid >> 2)) * N + nbase) + (tid & 3) * 32;
            __builtin_prefetch(pf, 0, 1);
        }
        #pragma unroll
        for (int i = 0; i < 2; ++i) {
            #pragma unroll
            for (int j = 0; j < 8; ++j)
                lB[cur][(n8Idx[i] + j) * LSTR + kIdx[i]] = bq[i].s[j];
        }

#if HAVE_TDM
        if (wave == 0) __builtin_amdgcn_s_wait_tensorcnt(0);   // cur A tile landed
#else
        stage_A_manual(&lA[cur][0], s.A, s.K, mbase, s.kb, tid);
#endif
        __syncthreads();   // cur A (TDM) + cur B visible to all waves

#if HAVE_TDM
        // issue next A tile DMA now: every wave is past reading lA[nxt]
        if (c + 1 < nch && wave == 0) {
            Chunk t = get_chunk(c + 1, n0, A0, B0, K0, A1, B1, K1);
            tdm_load_A(t.A, t.K, mbase, t.kb, ldsA[nxt]);
        }
#endif

        // ---- fragments + 8 WMMAs ----
        const int khalf = (lane >> 4) << 3;     // 0 or 8
        Frag af[2];
        #pragma unroll
        for (int at = 0; at < 2; ++at) {
            int r = (wave << 5) + (at << 4) + (lane & 15);
            const unsigned short* base = &lA[cur][r * LSTR];
            #pragma unroll
            for (int v = 0; v < 4; ++v) {
                af[at].u[v]     = *(const unsigned int*)(base + khalf + 2 * v);
                af[at].u[v + 4] = *(const unsigned int*)(base + 16 + khalf + 2 * v);
            }
        }
        #pragma unroll
        for (int nt = 0; nt < 4; ++nt) {
            Frag bfr;
            int r = (nt << 4) + (lane & 15);
            const unsigned short* base = &lB[cur][r * LSTR];
            #pragma unroll
            for (int v = 0; v < 4; ++v) {
                bfr.u[v]     = *(const unsigned int*)(base + khalf + 2 * v);
                bfr.u[v + 4] = *(const unsigned int*)(base + 16 + khalf + 2 * v);
            }
#if defined(__gfx1250__)
            acc[0][nt] = __builtin_amdgcn_wmma_f32_16x16x32_bf16(
                false, af[0].v, false, bfr.v, (short)0, acc[0][nt], false, false);
            acc[1][nt] = __builtin_amdgcn_wmma_f32_16x16x32_bf16(
                false, af[1].v, false, bfr.v, (short)0, acc[1][nt], false, false);
#endif
        }
        // next iteration stages into the other buffers; the barrier above protects reuse
    }

    // ---- epilogue: C layout -> col = lane&15, rows r + 8*(lane>>4) ----
    const int half = lane >> 4;
    const int col  = lane & 15;
    #pragma unroll
    for (int at = 0; at < 2; ++at) {
        const int mrow0 = mbase + (wave << 5) + (at << 4) + (half << 3);
        #pragma unroll
        for (int nt = 0; nt < 4; ++nt) {
            int ncol = nbase + (nt << 4) + col;
            float bv = 0.f;
            if (bias != nullptr && ncol < biasN) bv = bias[ncol];
            #pragma unroll
            for (int r = 0; r < 8; ++r) {
                float f = acc[at][nt][r] + bv;
                if (epilogue == 1) f = fmaxf(f, 0.f);
                else if (epilogue == 2) f = tanhf(f);
                size_t oi = (size_t)(mrow0 + r) * N + ncol;
                if (outF)  outF[oi] = f;
                if (outBf) outBf[oi] = f32_to_bf16(f);
            }
        }
    }
    (void)M;
}

// ---------------- softmax + categorical sample + state update (1 wave per row) ----------------
__global__ void sample_kernel(const float* __restrict__ logits_p, const float* __restrict__ logits_h,
                              const int* __restrict__ lens, const int* __restrict__ minTmax,
                              int* __restrict__ t_index, int* __restrict__ left,
                              int* __restrict__ out_paths, int* __restrict__ out_frames,
                              float* __restrict__ out_probs, float* __restrict__ out_scores,
                              int* __restrict__ out_left, int step) {
    const int wavesPerBlock = blockDim.x >> 5;
    const int row  = blockIdx.x * wavesPerBlock + (threadIdx.x >> 5);
    const int lane = threadIdx.x & 31;
    if (row >= NROWS) return;
    const float* lg = logits_p + (size_t)row * NP;

    float mx = -3.4e38f;
    for (int j = lane; j < VV; j += 32) mx = fmaxf(mx, lg[j]);
    #pragma unroll
    for (int o = 16; o > 0; o >>= 1) mx = fmaxf(mx, __shfl_xor(mx, o, 32));
    float sum = 0.f;
    for (int j = lane; j < VV; j += 32) sum += __expf(lg[j] - mx);
    #pragma unroll
    for (int o = 16; o > 0; o >>= 1) sum += __shfl_xor(sum, o, 32);

    unsigned long long s = splitmix64(0x1234567ull ^ ((unsigned long long)step * 4096ull + (unsigned long long)row));
    float u = (float)(s >> 40) * (1.0f / 16777216.0f);
    float target = u * sum;

    bool found = false; int idx = VV - 1; float pick = 0.f; float run = 0.f;
    for (int base = 0; base < NP; base += 32) {
        int j = base + lane;
        float e = (j < VV) ? __expf(lg[j] - mx) : 0.f;
        float sc = e;
        #pragma unroll
        for (int o = 1; o < 32; o <<= 1) { float t = __shfl_up(sc, o, 32); if (lane >= o) sc += t; }
        unsigned long long m = __ballot((run + sc) >= target && j < VV);
        if (m != 0ull) {
            int fl = __ffsll((unsigned long long)m) - 1;
            idx = base + fl;
            pick = __shfl(e, fl, 32);
            found = true;
            break;
        }
        run += __shfl(sc, 31, 32);
    }
    if (!found) pick = __expf(lg[VV - 1] - mx);
    float samp = pick / sum;
    float score = logits_h[(size_t)row * NP + idx];

    if (lane == 0) {
        int t = t_index[row];
        int l0 = left[2 * row + 0], l1 = left[2 * row + 1];
        size_t oi = (size_t)row * LL + step;
        out_paths[oi]  = idx;
        out_frames[oi] = t;
        out_probs[oi]  = samp;
        out_scores[oi] = score;
        out_left[2 * oi + 0] = l0;
        out_left[2 * oi + 1] = l1;

        int b = row >> 8;
        int tmax = lens[b];
        int tnew = t + 1;
        bool blank = (idx == 0);
        bool fin   = (tnew >= tmax);
        int mt = max(minTmax[0], 1);
        int refill = (int)(splitmix64(0xABCDEFull + (unsigned long long)step) % (unsigned long long)mt);
        int nl0 = blank ? l0 : l1;
        int nl1 = blank ? l1 : idx;
        if (fin) { nl0 = 0; nl1 = 0; tnew = refill; }
        left[2 * row + 0] = nl0;
        left[2 * row + 1] = nl1;
        t_index[row] = tnew;
    }
}

// ---------------- host orchestration ----------------
extern "C" void kernel_launch(void* const* d_in, const int* in_sizes, int n_in,
                              void* d_out, int out_size, void* d_ws, size_t ws_size,
                              hipStream_t stream) {
    const float* encoder_out = (const float*)d_in[0];
    const int*   enc_lens    = (const int*)d_in[1];
    const float* emb_p   = (const float*)d_in[2];
    const float* Wdec_p  = (const float*)d_in[3];
    const float* bdec_p  = (const float*)d_in[4];
    const float* emb_h   = (const float*)d_in[5];
    const float* Wdec_h  = (const float*)d_in[6];
    const float* bdec_h  = (const float*)d_in[7];
    const float* We_p    = (const float*)d_in[8];
    const float* Wd_p    = (const float*)d_in[9];
    const float* Wo_p    = (const float*)d_in[10];
    const float* bo_p    = (const float*)d_in[11];
    const float* We_h    = (const float*)d_in[12];
    const float* Wd_h    = (const float*)d_in[13];
    const float* Wo_h    = (const float*)d_in[14];
    const float* bo_h    = (const float*)d_in[15];

    int*   out_paths  = (int*)d_out;                       // B*P*L
    int*   out_frames = out_paths + NROWS * LL;            // B*P*L
    float* out_probs  = (float*)(out_frames + NROWS * LL); // B*P*L
    float* out_scores = out_probs + NROWS * LL;            // B*P*L
    int*   out_left   = (int*)(out_scores + NROWS * LL);   // B*P*L*CTX

    size_t off = 0;
    auto wsal = [&](size_t bytes) -> char* {
        char* p = (char*)d_ws + off;
        off += (bytes + 255) & ~(size_t)255;
        return p;
    };
    int* t_index = (int*)wsal(NROWS * 4);
    int* left    = (int*)wsal(NROWS * 2 * 4);
    int* minTmax = (int*)wsal(4);
    unsigned short* Wdec_p_bf = (unsigned short*)wsal((size_t)CTX * DD * NP * 2);
    unsigned short* Wdec_h_bf = (unsigned short*)wsal((size_t)CTX * DD * NP * 2);
    unsigned short* We_p_bf   = (unsigned short*)wsal((size_t)EE * NP * 2);
    unsigned short* Wd_p_bf   = (unsigned short*)wsal((size_t)DD * NP * 2);
    unsigned short* We_h_bf   = (unsigned short*)wsal((size_t)EE * NP * 2);
    unsigned short* Wd_h_bf   = (unsigned short*)wsal((size_t)DD * NP * 2);
    unsigned short* Wo_p_bf   = (unsigned short*)wsal((size_t)JJ * NP * 2);
    unsigned short* Wo_h_bf   = (unsigned short*)wsal((size_t)JJ * NP * 2);
    unsigned short* encX = (unsigned short*)wsal((size_t)NROWS * EE * 2);
    unsigned short* Xp   = (unsigned short*)wsal((size_t)NROWS * CTX * DD * 2);
    unsigned short* Xh   = (unsigned short*)wsal((size_t)NROWS * CTX * DD * 2);
    unsigned short* d_p  = (unsigned short*)wsal((size_t)NROWS * DD * 2);
    unsigned short* d_h  = (unsigned short*)wsal((size_t)NROWS * DD * 2);
    unsigned short* z_p  = (unsigned short*)wsal((size_t)NROWS * JJ * 2);
    unsigned short* z_h  = (unsigned short*)wsal((size_t)NROWS * JJ * 2);
    float* logits_p = (float*)wsal((size_t)NROWS * NP * 4);
    float* logits_h = (float*)wsal((size_t)NROWS * NP * 4);

    init_kernel<<<(NROWS + 255) / 256, 256, 0, stream>>>(enc_lens, t_index, left, minTmax);

    auto conv = [&](const float* src, unsigned short* dst, int rows, int cols, int dstCols) {
        int total = rows * dstCols;
        convert_kernel<<<(total + 255) / 256, 256, 0, stream>>>(src, dst, rows, cols, dstCols);
    };
    conv(Wdec_p, Wdec_p_bf, CTX * DD, NP, NP);
    conv(Wdec_h, Wdec_h_bf, CTX * DD, NP, NP);
    conv(We_p,   We_p_bf,   EE, NP, NP);
    conv(Wd_p,   Wd_p_bf,   DD, NP, NP);
    conv(We_h,   We_h_bf,   EE, NP, NP);
    conv(Wd_h,   Wd_h_bf,   DD, NP, NP);
    conv(Wo_p,   Wo_p_bf,   JJ, VV, NP);   // pad 500 -> 512 cols with zeros
    conv(Wo_h,   Wo_h_bf,   JJ, VV, NP);

    dim3 gg(NROWS / BM, NP / BN);   // (32, 8)

    for (int step = 0; step < LL; ++step) {
        gather_kernel<<<NROWS, 128, 0, stream>>>(encoder_out, t_index, left,
                                                 emb_p, emb_h, encX, Xp, Xh);
        wmma_gemm_kernel<<<gg, 128, 0, stream>>>(Xp, Wdec_p_bf, CTX * DD,
                                                 nullptr, nullptr, 0,
                                                 bdec_p, NP, NROWS, NP, d_p, nullptr, 1);
        wmma_gemm_kernel<<<gg, 128, 0, stream>>>(Xh, Wdec_h_bf, CTX * DD,
                                                 nullptr, nullptr, 0,
                                                 bdec_h, NP, NROWS, NP, d_h, nullptr, 1);
        wmma_gemm_kernel<<<gg, 128, 0, stream>>>(encX, We_p_bf, EE,
                                                 d_p, Wd_p_bf, DD,
                                                 nullptr, 0, NROWS, NP, z_p, nullptr, 2);
        wmma_gemm_kernel<<<gg, 128, 0, stream>>>(encX, We_h_bf, EE,
                                                 d_h, Wd_h_bf, DD,
                                                 nullptr, 0, NROWS, NP, z_h, nullptr, 2);
        wmma_gemm_kernel<<<gg, 128, 0, stream>>>(z_p, Wo_p_bf, JJ,
                                                 nullptr, nullptr, 0,
                                                 bo_p, VV, NROWS, NP, nullptr, logits_p, 0);
        wmma_gemm_kernel<<<gg, 128, 0, stream>>>(z_h, Wo_h_bf, JJ,
                                                 nullptr, nullptr, 0,
                                                 bo_h, VV, NROWS, NP, nullptr, logits_h, 0);
        sample_kernel<<<NROWS / 8, 256, 0, stream>>>(logits_p, logits_h, enc_lens, minTmax,
                                                     t_index, left,
                                                     out_paths, out_frames, out_probs,
                                                     out_scores, out_left, step);
    }
    (void)in_sizes; (void)n_in; (void)out_size; (void)ws_size;
}
